// GatedAttentionPooling_46815143526542
// MI455X (gfx1250) — matrix operand order repaired and checked by reading, not compiled, not run on hardware
//
#include <hip/hip_runtime.h>
#include <hip/hip_bf16.h>
#include <stdint.h>

// ---------------------------------------------------------------------------
// GatedAttentionPooling for MI455X (gfx1250, wave32, WMMA bf16)
//   z[b,:] = sum_n in seg(b)  softmax_seg( (tanh(xW1^T) * softmax_H(xW2^T)) W3^T )_n * x[n,:]
// ---------------------------------------------------------------------------

#define D_DIM 1024
#define H_DIM 512
#define B_SEG 512
#define WAVES_PER_BLOCK 4
#define ROWS_PER_WAVE 16
#define ROWS_PER_BLOCK (WAVES_PER_BLOCK * ROWS_PER_WAVE)

typedef __attribute__((ext_vector_type(16))) __bf16 v16bf;
typedef __attribute__((ext_vector_type(8)))  __bf16 v8bf;
typedef __attribute__((ext_vector_type(8)))  float  v8f;

// ------------------------- helpers -----------------------------------------

__device__ __forceinline__ unsigned fmono(float f) {
    unsigned u = __float_as_uint(f);
    return (u & 0x80000000u) ? ~u : (u | 0x80000000u);
}
__device__ __forceinline__ float fmono_inv(unsigned m) {
    unsigned u = (m & 0x80000000u) ? (m & 0x7FFFFFFFu) : ~m;
    return __uint_as_float(u);
}

// branchless tanh: (e^2y - 1) / (e^2y + 1), clamped for overflow safety
__device__ __forceinline__ float fast_tanh(float y) {
    float yc = fminf(fmaxf(y, -15.0f), 15.0f);
    float t  = __expf(2.0f * yc);
    return (t - 1.0f) * __builtin_amdgcn_rcpf(t + 1.0f);
}

__device__ __forceinline__ int lower_bound_i64(const long long* __restrict__ b,
                                               int n, long long key) {
    int lo = 0, hi = n;
    while (lo < hi) {
        int mid = (lo + hi) >> 1;
        if (b[mid] < key) lo = mid + 1; else hi = mid;
    }
    return lo;
}

// ------------------------- kernel 0: weight convert fp32 -> bf16 -----------

__global__ void gap_wcvt_kernel(const float* __restrict__ W1,
                                const float* __restrict__ W2,
                                __hip_bfloat16* __restrict__ W1b,
                                __hip_bfloat16* __restrict__ W2b, int n) {
    int i = blockIdx.x * blockDim.x + threadIdx.x;
    if (i < n) {
        W1b[i] = __float2bfloat16(W1[i]);
        W2b[i] = __float2bfloat16(W2[i]);
    }
}

// ------------------------- kernel 0b: init segment scratch -----------------

__global__ void gap_seg_init_kernel(unsigned* __restrict__ smax,
                                    float* __restrict__ ssum) {
    int i = blockIdx.x * blockDim.x + threadIdx.x;
    if (i < B_SEG) { smax[i] = 0u; ssum[i] = 0.0f; }   // fmono_inv(0) below all reals
}

// ------------------------- kernel 1: fused GEMM + gate ---------------------
// One wave owns 16 rows of x. Pipeline:
//   1) async-LDS DMA: 16x1024 f32 tile (64 KB) global -> LDS  (ASYNCcnt)
//   2) in-place convert to bf16 (front 32 KB of the same region; DS in-order)
//   3) online softmax over H, 32 h-values per pass, 4 WMMA bf16 per k-step

__global__ __launch_bounds__(WAVES_PER_BLOCK * 32)
void gap_gemm_alpha_kernel(const float* __restrict__ x,
                           const __hip_bfloat16* __restrict__ W1b,
                           const __hip_bfloat16* __restrict__ W2b,
                           const float* __restrict__ W3,
                           float* __restrict__ aOut, int N) {
    extern __shared__ float smem[];                  // WAVES*16*1024 f32 = 256 KB
    const int lane = threadIdx.x & 31;
    const int wave = threadIdx.x >> 5;
    const int half = lane >> 4;                      // K-half / row-half select
    const int lg   = lane & 15;                      // group lane (M for A, N for B/C)

    const int tile = blockIdx.x * WAVES_PER_BLOCK + wave;
    const int m0   = tile * ROWS_PER_WAVE;
    if (m0 >= N) return;

    float*          xt  = smem + wave * (ROWS_PER_WAVE * D_DIM);
    __hip_bfloat16* xtb = (__hip_bfloat16*)xt;       // bf16 alias, front 32 KB

    // ---- 1) stage tile global -> LDS via async-LDS DMA --------------------
    {
        const uint64_t gbase = (uint64_t)(uintptr_t)(x + (size_t)m0 * D_DIM);
        const uint32_t lbase = (uint32_t)(uintptr_t)xt;
        #pragma unroll 4
        for (int i = 0; i < 128; ++i) {              // 128 * 32 lanes * 16 B = 64 KB
            uint32_t off   = (uint32_t)(i * 32 + lane) * 16u;
            uint32_t laddr = lbase + off;
            uint64_t gaddr = gbase + off;
            asm volatile("global_load_async_to_lds_b128 %0, %1, off"
                         :: "v"(laddr), "v"(gaddr) : "memory");
        }
        asm volatile("s_wait_asynccnt 0" ::: "memory");
    }

    // ---- 2) in-place f32 -> bf16 compaction (reads outrun writes) ---------
    // group g: loads cover bytes [4096g, 4096g+4096), stores [2048g, 2048g+2048)
    for (int g = 0; g < 16; ++g) {
        float4 c[8];
        #pragma unroll
        for (int j = 0; j < 4; ++j) {
            int idx = g * 128 + j * 32 + lane;       // 8 floats per idx
            c[2*j]   = *(const float4*)(xt + idx * 8);
            c[2*j+1] = *(const float4*)(xt + idx * 8 + 4);
        }
        #pragma unroll
        for (int j = 0; j < 4; ++j) {
            int idx = g * 128 + j * 32 + lane;
            v8bf p;
            p[0] = (__bf16)c[2*j].x;   p[1] = (__bf16)c[2*j].y;
            p[2] = (__bf16)c[2*j].z;   p[3] = (__bf16)c[2*j].w;
            p[4] = (__bf16)c[2*j+1].x; p[5] = (__bf16)c[2*j+1].y;
            p[6] = (__bf16)c[2*j+1].z; p[7] = (__bf16)c[2*j+1].w;
            *(v8bf*)(xtb + idx * 8) = p;
        }
    }

    // ---- 3) online softmax over H ----------------------------------------
    float m_run[8], d_run[8], num[8];
    #pragma unroll
    for (int r = 0; r < 8; ++r) { m_run[r] = -3.0e38f; d_run[r] = 0.0f; num[r] = 0.0f; }

    // A 16x32 bf16 layout: M = lane&15; lanes<16 K=0..15, lanes>=16 K=16..31
    const __hip_bfloat16* arow = xtb + lg * D_DIM;

    for (int h0 = 0; h0 < H_DIM; h0 += 32) {         // two 16-wide h-subtiles
        v8f accU0 = {}, accV0 = {}, accU1 = {}, accV1 = {};
        // B 32x16 layout: column n = lane&15; lanes<16 K=0..15, lanes>=16 K=16..31
        const __hip_bfloat16* b1p0 = W1b + (size_t)(h0 + lg) * D_DIM + half * 16;
        const __hip_bfloat16* b2p0 = W2b + (size_t)(h0 + lg) * D_DIM + half * 16;
        const __hip_bfloat16* b1p1 = b1p0 + 16 * D_DIM;
        const __hip_bfloat16* b2p1 = b2p0 + 16 * D_DIM;
        if (h0 + 32 < H_DIM) {                       // pull next h-tile of weights
            __builtin_prefetch(b1p0 + 32 * D_DIM, 0, 0);
            __builtin_prefetch(b2p0 + 32 * D_DIM, 0, 0);
        }
        #pragma unroll 2
        for (int k0 = 0; k0 < D_DIM; k0 += 32) {
            v8bf lo = *(const v8bf*)(arow + k0 + half * 16);       // ds_load_b128
            v8bf hi = *(const v8bf*)(arow + k0 + half * 16 + 8);   // ds_load_b128
            v16bf af = __builtin_shufflevector(lo, hi,
                        0,1,2,3,4,5,6,7,8,9,10,11,12,13,14,15);
            v16bf b10 = *(const v16bf*)(b1p0 + k0);
            v16bf b20 = *(const v16bf*)(b2p0 + k0);
            v16bf b11 = *(const v16bf*)(b1p1 + k0);
            v16bf b21 = *(const v16bf*)(b2p1 + k0);
            accU0 = __builtin_amdgcn_wmma_f32_16x16x32_bf16(
                        false, af, false, b10, (short)0, accU0, false, false);
            accV0 = __builtin_amdgcn_wmma_f32_16x16x32_bf16(
                        false, af, false, b20, (short)0, accV0, false, false);
            accU1 = __builtin_amdgcn_wmma_f32_16x16x32_bf16(
                        false, af, false, b11, (short)0, accU1, false, false);
            accV1 = __builtin_amdgcn_wmma_f32_16x16x32_bf16(
                        false, af, false, b21, (short)0, accV1, false, false);
        }

        // C/D layout: VGPR r -> row (half*8 + r), column n = lane&15
        const float w3a = W3[h0 + lg];
        const float w3b = W3[h0 + 16 + lg];
        #pragma unroll
        for (int r = 0; r < 8; ++r) {
            float u0 = fast_tanh(accU0[r]);
            float u1 = fast_tanh(accU1[r]);
            float v0 = accV0[r], v1 = accV1[r];
            float mx = fmaxf(v0, v1);                // row max over 32 h values
            mx = fmaxf(mx, __shfl_xor(mx, 1, 16));
            mx = fmaxf(mx, __shfl_xor(mx, 2, 16));
            mx = fmaxf(mx, __shfl_xor(mx, 4, 16));
            mx = fmaxf(mx, __shfl_xor(mx, 8, 16));
            float nm  = fmaxf(m_run[r], mx);
            float sc  = __expf(m_run[r] - nm);       // rescale old partial sums
            float ev0 = __expf(v0 - nm);
            float ev1 = __expf(v1 - nm);
            float se  = ev0 + ev1;
            float st  = u0 * ev0 * w3a + u1 * ev1 * w3b;
            se += __shfl_xor(se, 1, 16);  st += __shfl_xor(st, 1, 16);
            se += __shfl_xor(se, 2, 16);  st += __shfl_xor(st, 2, 16);
            se += __shfl_xor(se, 4, 16);  st += __shfl_xor(st, 4, 16);
            se += __shfl_xor(se, 8, 16);  st += __shfl_xor(st, 8, 16);
            d_run[r] = d_run[r] * sc + se;
            num[r]   = num[r]   * sc + st;
            m_run[r] = nm;
        }
    }

    if (lg == 0) {                                   // lanes 0 & 16 own rows 0-7 / 8-15
        #pragma unroll
        for (int r = 0; r < 8; ++r)
            aOut[m0 + half * 8 + r] = num[r] / d_run[r];
    }
}

// ------------------------- kernel 2: segment max ---------------------------

__global__ void gap_seg_max_kernel(const float* __restrict__ a,
                                   const long long* __restrict__ batch,
                                   unsigned* __restrict__ smax, int N) {
    int n = blockIdx.x * blockDim.x + threadIdx.x;
    if (n < N) atomicMax(&smax[(int)batch[n]], fmono(a[n]));
}

// ------------------------- kernel 3: segment exp-sum -----------------------

__global__ void gap_seg_expsum_kernel(const float* __restrict__ a,
                                      const long long* __restrict__ batch,
                                      const unsigned* __restrict__ smax,
                                      float* __restrict__ e,
                                      float* __restrict__ ssum, int N) {
    int n = blockIdx.x * blockDim.x + threadIdx.x;
    if (n < N) {
        int b = (int)batch[n];
        float ev = __expf(a[n] - fmono_inv(smax[b]));
        e[n] = ev;
        atomicAdd(&ssum[b], ev);
    }
}

// ------------------------- kernel 4: weighted segment pooling --------------
// One block per segment; batch is sorted so rows are contiguous. Accumulate a
// full D=1024 row in registers (float4/thread), single store, no atomics.

__global__ __launch_bounds__(256)
void gap_seg_pool_kernel(const float* __restrict__ x,
                         const long long* __restrict__ batch,
                         const float* __restrict__ e,
                         const float* __restrict__ ssum,
                         float* __restrict__ z, int N) {
    __shared__ int rng[2];
    const int b = blockIdx.x;
    if (threadIdx.x == 0) {
        rng[0] = lower_bound_i64(batch, N, (long long)b);
        rng[1] = lower_bound_i64(batch, N, (long long)b + 1);
    }
    __syncthreads();
    const int lo = rng[0], hi = rng[1];
    const float inv = (hi > lo) ? 1.0f / ssum[b] : 0.0f;

    const float4* x4 = (const float4*)x;
    const int t = threadIdx.x;                       // 256 threads * float4 = 1024 cols
    float4 acc = make_float4(0.f, 0.f, 0.f, 0.f);
    for (int n = lo; n < hi; ++n) {
        float  w  = e[n] * inv;
        float4 xv = x4[(size_t)n * (D_DIM / 4) + t];
        acc.x = fmaf(xv.x, w, acc.x);
        acc.y = fmaf(xv.y, w, acc.y);
        acc.z = fmaf(xv.z, w, acc.z);
        acc.w = fmaf(xv.w, w, acc.w);
    }
    ((float4*)z)[(size_t)b * (D_DIM / 4) + t] = acc;
}

// ------------------------- launch ------------------------------------------

extern "C" void kernel_launch(void* const* d_in, const int* in_sizes, int n_in,
                              void* d_out, int out_size, void* d_ws, size_t ws_size,
                              hipStream_t stream) {
    const float*     x     = (const float*)d_in[0];
    const long long* batch = (const long long*)d_in[1];
    const float*     W1    = (const float*)d_in[2];
    const float*     W2    = (const float*)d_in[3];
    const float*     W3    = (const float*)d_in[4];
    float*           z     = (float*)d_out;

    const int N  = in_sizes[1];                      // 262144
    const int HD = H_DIM * D_DIM;                    // 524288

    // workspace carve-up (~4.2 MB total)
    char* ws = (char*)d_ws;
    __hip_bfloat16* W1b = (__hip_bfloat16*)ws;  ws += (size_t)HD * 2;
    __hip_bfloat16* W2b = (__hip_bfloat16*)ws;  ws += (size_t)HD * 2;
    float*    aArr = (float*)ws;                ws += (size_t)N * 4;
    float*    eArr = (float*)ws;                ws += (size_t)N * 4;
    unsigned* smax = (unsigned*)ws;             ws += (size_t)B_SEG * 4;
    float*    ssum = (float*)ws;                ws += (size_t)B_SEG * 4;

    gap_wcvt_kernel<<<(HD + 255) / 256, 256, 0, stream>>>(W1, W2, W1b, W2b, HD);
    gap_seg_init_kernel<<<(B_SEG + 255) / 256, 256, 0, stream>>>(smax, ssum);

    const size_t lds_bytes = (size_t)WAVES_PER_BLOCK * ROWS_PER_WAVE * D_DIM * 4; // 256 KB
    const int grid1 = (N + ROWS_PER_BLOCK - 1) / ROWS_PER_BLOCK;
    gap_gemm_alpha_kernel<<<grid1, WAVES_PER_BLOCK * 32, lds_bytes, stream>>>(
        x, W1b, W2b, W3, aArr, N);

    gap_seg_max_kernel<<<(N + 255) / 256, 256, 0, stream>>>(aArr, batch, smax, N);
    gap_seg_expsum_kernel<<<(N + 255) / 256, 256, 0, stream>>>(aArr, batch, smax,
                                                               eArr, ssum, N);
    gap_seg_pool_kernel<<<B_SEG, 256, 0, stream>>>(x, batch, eArr, ssum, z, N);
}
